// MemoryLocal_26362509263545
// MI455X (gfx1250) — matrix operand order repaired and checked by reading, not compiled, not run on hardware
//
#include <hip/hip_runtime.h>

typedef __attribute__((ext_vector_type(16))) _Float16 v16h;
typedef __attribute__((ext_vector_type(8)))  _Float16 v8h;
typedef __attribute__((ext_vector_type(2)))  _Float16 v2h;
typedef __attribute__((ext_vector_type(8)))  float    v8f;

#define B_    2
#define DE    128
#define DO    256
#define T_    4
#define H_    120
#define W_    216
#define HW    (H_*W_)
#define KT    36            // 9 taps * T
#define TILE  14            // output pixels per 16-wide GEMM tile (band fits)
#define NTILE 16            // ceil(216/14)
#define MOPAD 24            // halves per staged m_out row (48B, 16B aligned)

__device__ __forceinline__ int clampi(int x, int lo, int hi) {
  return x < lo ? lo : (x > hi ? hi : x);
}

// ---------------------------------------------------------------------------
// Kernel A: correlation (WMMA f16->f32) + softmax over 36, writes p.
// One wave per (b, h, 14-pixel tile). Branchless OOB (clamp + select),
// base-pointer addressing so channel offsets fold into 24-bit immediates.
// ---------------------------------------------------------------------------
__global__ __launch_bounds__(32) void corr_softmax_kernel(
    const float* __restrict__ q, const float* __restrict__ m_in,
    float* __restrict__ p_out)
{
  __shared__ float corr_lds[KT * 16];

  const int lane = threadIdx.x;
  const int hi   = lane >> 4;
  const int m    = lane & 15;
  const int w0   = blockIdx.x * TILE;
  const int h    = blockIdx.y;
  const int b    = blockIdx.z;

  // ---- A operand: 16 pixels (M) x 128 channels (K); channel = ch*32+16*hi+k0
  v16h aq[4];
  {
    const int  wp  = w0 + m;                    // M = lane&15
    const bool qok = (wp < W_);
    const int  wpc = qok ? wp : (W_ - 1);
    const float* baseA =
        q + (size_t)b * DE * HW + (size_t)hi * 16 * HW + (size_t)h * W_ + wpc;
#pragma unroll
    for (int ch = 0; ch < 4; ++ch) {
      const float* pA = baseA + (size_t)ch * 32 * HW;
      float f[16];
#pragma unroll
      for (int v = 0; v < 8; ++v) {
        const int k0 = 2 * v;   // kmapA - 16*hi
        f[2*v]     = pA[(size_t)k0 * HW];
        f[2*v + 1] = pA[(size_t)(k0 + 1) * HW];
      }
#pragma unroll
      for (int e = 0; e < 16; ++e)
        aq[ch][e] = (_Float16)(qok ? f[e] : 0.f);
    }
  }

#pragma unroll 1
  for (int t = 0; t < T_; ++t) {
#pragma unroll
    for (int dy = 0; dy < 3; ++dy) {
      const int  row   = h + dy - 1;
      const int  wp    = w0 - 1 + m;            // N = lane&15
      const bool ok    = (row >= 0) && (row < H_) && (wp >= 0) && (wp < W_);
      const int  rowc  = clampi(row, 0, H_ - 1);
      const int  wpc   = clampi(wp, 0, W_ - 1);
      // channel = ch*32 + 16*hi + 2v(+1); fold b,t,hi,row,wp into base
      const float* baseB = m_in + (size_t)b * DE * T_ * HW + (size_t)t * HW +
                           (size_t)hi * 16 * T_ * HW + (size_t)rowc * W_ + wpc;
      v16h bm[4];
#pragma unroll
      for (int ch = 0; ch < 4; ++ch) {
        const float* pB = baseB + (size_t)ch * 32 * T_ * HW;
        float f[16];
#pragma unroll
        for (int v = 0; v < 8; ++v) {
          f[2*v]     = pB[(size_t)(2 * v) * T_ * HW];
          f[2*v + 1] = pB[(size_t)(2 * v + 1) * T_ * HW];
        }
#pragma unroll
        for (int e = 0; e < 16; ++e)
          bm[ch][e] = (_Float16)(ok ? f[e] : 0.f);
      }

      v8f acc = {};
#pragma unroll
      for (int ch = 0; ch < 4; ++ch)
        acc = __builtin_amdgcn_wmma_f32_16x16x32_f16(
            false, aq[ch], false, bm[ch], (short)0, acc, false, false);

      // Extract the 3 diagonals: element (M,N), M=r+8*hi, N=m; dx = N-M
#pragma unroll
      for (int r = 0; r < 8; ++r) {
        const int M = r + 8 * hi;
        const int d = m - M;
        if (d >= 0 && d <= 2 && M < TILE) {
          corr_lds[((dy * 3 + d) * T_ + t) * 16 + M] = acc[r] * (1.0f / (float)DE);
        }
      }
    }
  }

  __syncthreads();

  // Softmax across all 36 (k,t) per pixel; lanes 0..13 handle one pixel each
  const int i = lane;
  if (i < TILE && (w0 + i) < W_) {
    float vals[KT];
    float mx = -3.4e38f;
#pragma unroll
    for (int kt = 0; kt < KT; ++kt) {
      vals[kt] = corr_lds[kt * 16 + i];
      mx = fmaxf(mx, vals[kt]);
    }
    float s = 0.f;
#pragma unroll
    for (int kt = 0; kt < KT; ++kt) {
      vals[kt] = __expf(vals[kt] - mx);
      s += vals[kt];
    }
    const float inv = 1.0f / s;
#pragma unroll
    for (int kt = 0; kt < KT; ++kt) {
      p_out[((size_t)(b * KT + kt) * H_ + h) * W_ + w0 + i] = vals[kt] * inv;
    }
  }
}

// ---------------------------------------------------------------------------
// Kernel B helpers: software-pipelined staging of one dy-row of m_out
// (4t x 256c x 18j) through registers into LDS.
// ---------------------------------------------------------------------------
__device__ __forceinline__ void stage_load(const float* __restrict__ m_out,
                                           int b, int w0, int row, bool rowok,
                                           int tid, float (&rg)[36]) {
  const int rowc = rowok ? row : 0;
#pragma unroll
  for (int it = 0; it < 36; ++it) {
    const int idx = tid + it * 512;                 // 36*512 == 4*256*18
    const int t   = idx / (DO * 18);
    const int rem = idx % (DO * 18);
    const int c   = rem / 18;
    const int jj  = rem % 18;
    const int wp  = w0 - 1 + jj;
    const bool ok = rowok && (wp >= 0) && (wp < W_);
    const int wpc = clampi(wp, 0, W_ - 1);
    const float fv =
        m_out[((((size_t)b * DO + c) * T_ + t) * H_ + rowc) * W_ + wpc];
    rg[it] = ok ? fv : 0.f;
  }
}

__device__ __forceinline__ void stage_store(_Float16* mo_lds, int tid,
                                            const float (&rg)[36]) {
#pragma unroll
  for (int it = 0; it < 36; ++it) {
    const int idx = tid + it * 512;
    const int t   = idx / (DO * 18);
    const int rem = idx % (DO * 18);
    const int c   = rem / 18;
    const int jj  = rem % 18;
    mo_lds[(t * DO + c) * MOPAD + jj] = (_Float16)rg[it];
  }
}

// ---------------------------------------------------------------------------
// Kernel B: aggregation as banded WMMA GEMM. 512 threads = 16 waves; each
// wave owns 16 channels of Do. m_out rows staged f16 into LDS (48B stride ->
// ds_load_b128 gathers), with dy+1's global loads overlapped with dy's WMMAs.
// ---------------------------------------------------------------------------
__global__ __launch_bounds__(512) void aggregate_kernel(
    const float* __restrict__ m_out, const float* __restrict__ p_in,
    float* __restrict__ mem_out)
{
  __shared__ float p_lds[KT * 16];
  __shared__ __align__(16) _Float16 mo_lds[T_ * DO * MOPAD];

  const int tid  = threadIdx.x;
  const int lane = tid & 31;
  const int wave = tid >> 5;
  const int hi   = lane >> 4;
  const int m    = lane & 15;
  const int c0   = wave * 16;
  const int w0   = blockIdx.x * TILE;
  const int h    = blockIdx.y;
  const int b    = blockIdx.z;

  // Softmax weights for this tile, branchless (slots i>=14 or OOB -> 0)
  for (int idx = tid; idx < KT * 16; idx += 512) {
    const int kt = idx >> 4;
    const int i  = idx & 15;
    const bool ok = (i < TILE) && (w0 + i < W_);
    const int  ic = ok ? i : 0;
    const float fv = p_in[((size_t)(b * KT + kt) * H_ + h) * W_ + w0 + ic];
    p_lds[idx] = ok ? fv : 0.f;
  }

  v8f accD = {};
  float rg[2][36];
  stage_load(m_out, b, w0, h - 1, (h - 1) >= 0, tid, rg[0]);

#pragma unroll
  for (int dy = 0; dy < 3; ++dy) {
    stage_store(mo_lds, tid, rg[dy & 1]);
    __syncthreads();                 // staging (and p_lds) visible to all
    if (dy < 2) {
      const int nrow = h + dy;       // (dy+1) - 1, always >= 0
      stage_load(m_out, b, w0, nrow, nrow < H_, tid, rg[(dy + 1) & 1]);
    }

#pragma unroll
    for (int t = 0; t < T_; ++t) {
      // A: M=16 channels, K=j. One ds_load_b128 covers slots v0..3
      // (K = 8*hi .. 8*hi+7); K=16,17 tail via ds_load_b32 (hi==0 lanes).
      const _Float16* rowp = &mo_lds[(t * DO + c0 + m) * MOPAD];
      const v8h lo  = *(const v8h*)(rowp + hi * 8);
      const v2h tl  = *(const v2h*)(rowp + 16);
      v16h am;
#pragma unroll
      for (int e = 0; e < 8; ++e) am[e] = lo[e];
      am[8] = hi ? (_Float16)0.f : tl[0];
      am[9] = hi ? (_Float16)0.f : tl[1];
#pragma unroll
      for (int e = 10; e < 16; ++e) am[e] = (_Float16)0.f;

      // B: banded weights W[j,i] = p[(dy, j-i), t, i] for j-i in {0,1,2}
      const int i = m;
      const _Float16 p0 = (_Float16)p_lds[((dy * 3 + 0) * T_ + t) * 16 + i];
      const _Float16 p1 = (_Float16)p_lds[((dy * 3 + 1) * T_ + t) * 16 + i];
      const _Float16 p2 = (_Float16)p_lds[((dy * 3 + 2) * T_ + t) * 16 + i];
      v16h bw;
#pragma unroll
      for (int v = 0; v < 8; ++v) {
        const int k  = (v < 4) ? (8 * hi + 2 * v) : (16 + 8 * hi + 2 * (v - 4));
        const int d0 = k - i;
        const int d1 = k + 1 - i;
        bw[2*v]     = (d0 == 0) ? p0 : (d0 == 1) ? p1 : (d0 == 2) ? p2 : (_Float16)0.f;
        bw[2*v + 1] = (d1 == 0) ? p0 : (d1 == 1) ? p1 : (d1 == 2) ? p2 : (_Float16)0.f;
      }
      accD = __builtin_amdgcn_wmma_f32_16x16x32_f16(
          false, am, false, bw, (short)0, accD, false, false);
    }
    __syncthreads();                 // compute done before LDS overwrite
  }

  // D layout: VGPR r -> (M=r+8*hi channel, N=m pixel); coalesced stores
#pragma unroll
  for (int r = 0; r < 8; ++r) {
    const int c = c0 + r + 8 * hi;
    if (m < TILE && (w0 + m) < W_) {
      mem_out[(((size_t)b * DO + c) * H_ + h) * W_ + w0 + m] = accD[r];
    }
  }
}

// ---------------------------------------------------------------------------
extern "C" void kernel_launch(void* const* d_in, const int* in_sizes, int n_in,
                              void* d_out, int out_size, void* d_ws, size_t ws_size,
                              hipStream_t stream) {
  (void)in_sizes; (void)n_in; (void)d_ws; (void)ws_size; (void)out_size;
  const float* m_in  = (const float*)d_in[0];
  const float* m_out = (const float*)d_in[1];
  const float* q_in  = (const float*)d_in[2];

  float* mem = (float*)d_out;                                   // (B,Do,H,W)
  float* p   = mem + (size_t)B_ * DO * H_ * W_;                 // (B,9,T,H,W)

  dim3 grid(NTILE, H_, B_);
  corr_softmax_kernel<<<grid, 32, 0, stream>>>(q_in, m_in, p);
  aggregate_kernel<<<grid, 512, 0, stream>>>(m_out, p, mem);
}
